// GraphStructureLearner_90606630076479
// MI455X (gfx1250) — matrix operand order repaired and checked by reading, not compiled, not run on hardware
//
#include <hip/hip_runtime.h>
#include <cstdint>
#include <cstddef>

typedef __attribute__((ext_vector_type(16))) __bf16 v16bf;
typedef __attribute__((ext_vector_type(8)))  float  v8f;
typedef __attribute__((ext_vector_type(4)))  float  f32x4;

#define HIDDEN 256
#define NNODE  4096
#define NBATCH 4
#define NEG_INF (-3.4e38f)

// ---------------------------------------------------------------------------
// Kernel 1: q = X @ Wq^T + bq ; k = X @ Wk^T + bk   (bf16 outputs in ws)
// One block = 16-row strip of the flattened [B*N, 256] X. 8 waves, 32 tile-jobs.
// X tile is staged into LDS with CDNA5 async-to-LDS copies (ASYNCcnt path).
// ---------------------------------------------------------------------------
__global__ void proj_kernel(const float* __restrict__ X,
                            const float* __restrict__ Wq, const float* __restrict__ bq,
                            const float* __restrict__ Wk, const float* __restrict__ bk,
                            __bf16* __restrict__ qb, __bf16* __restrict__ kb)
{
    __shared__ float  xstage[16 * HIDDEN];  // 16 KB f32 staging (async dest)
    __shared__ __bf16 xb[16 * HIDDEN];      // 8 KB bf16 operand tile
    const int row0 = blockIdx.x * 16;
    const int tid  = threadIdx.x;

    // Async copy: 4096 floats, 256 threads x 4 x b128 (16B per lane per op).
    {
        const float* gsrc = X + (size_t)row0 * HIDDEN;
#pragma unroll
        for (int p = 0; p < 4; ++p) {
            const int idx = tid * 4 + p * 1024;               // float index
            const uint32_t ldsa = (uint32_t)(uintptr_t)(&xstage[idx]);
            const float*   ga   = gsrc + idx;
            asm volatile("global_load_async_to_lds_b128 %0, %1, off"
                         :: "v"(ldsa), "v"(ga) : "memory");
        }
        asm volatile("s_wait_asynccnt 0x0" ::: "memory");
    }
    __syncthreads();
    for (int i = tid; i < 16 * HIDDEN; i += 256)
        xb[i] = (__bf16)xstage[i];
    __syncthreads();

    const int lane = tid & 31;
    const int wave = tid >> 5;
    const int m    = lane & 15;
    const int khalf = (lane >> 4) * 8;     // A-fragment K base (ISA 16-bit A layout)
    const int dh    = (lane >> 4) * 8;     // B-fragment K base
    const int mof   = (lane >> 4) * 8;     // C row offset

    // A fragments (shared across all jobs of this block)
    v16bf afr[8];
    const __bf16* xrow = xb + m * HIDDEN;
#pragma unroll
    for (int kt = 0; kt < 8; ++kt) {
        const int k0 = kt * 32 + khalf;
#pragma unroll
        for (int h = 0; h < 8; ++h) afr[kt][h]     = xrow[k0 + h];
#pragma unroll
        for (int h = 0; h < 8; ++h) afr[kt][8 + h] = xrow[k0 + 16 + h];
    }

    for (int jj = 0; jj < 4; ++jj) {
        const int job = wave + jj * 8;       // 0..31
        const int mat = job >> 4;            // 0 = q, 1 = k
        const int ct  = job & 15;            // column tile of 16
        const float* W    = mat ? Wk : Wq;
        const float* bias = mat ? bk : bq;
        __bf16*      outp = mat ? kb : qb;

        const int e = ct * 16 + m;                        // output column
        const float* wrow = W + (size_t)e * HIDDEN;       // W[e][:] contiguous
        v8f acc = {};
#pragma unroll
        for (int kt = 0; kt < 8; ++kt) {
            v16bf bfr;
#pragma unroll
            for (int h = 0; h < 16; ++h)
                bfr[h] = (__bf16)wrow[kt * 32 + dh + (h & 7) + ((h >> 3) << 4)];
            acc = __builtin_amdgcn_wmma_f32_16x16x32_bf16(
                false, afr[kt], false, bfr, (short)0, acc, false, false);
        }
        const float bv = bias[e];
#pragma unroll
        for (int v = 0; v < 8; ++v)
            outp[(size_t)(row0 + v + mof) * HIDDEN + e] = (__bf16)(acc[v] + bv);
    }
}

// ---------------------------------------------------------------------------
// Per-row top-8 + masked-softmax write.  scores row is 4096 f32 in LDS.
// ---------------------------------------------------------------------------
__device__ __forceinline__ void topk_row(const float* __restrict__ srow, int lane,
                                         int g, int* __restrict__ top_idx,
                                         float* __restrict__ top_val)
{
    float tv[8]; int tix[8];
#pragma unroll
    for (int j = 0; j < 8; ++j) { tv[j] = NEG_INF; tix[j] = 0; }

    // per-lane sorted (descending) top-8 over a 128-element stride slice
    for (int i = lane; i < NNODE; i += 32) {
        const float x = srow[i];
        if (x > tv[7]) {
            tv[7] = x; tix[7] = i;
#pragma unroll
            for (int j = 7; j > 0; --j) {
                if (tv[j] > tv[j - 1]) {
                    float a = tv[j]; tv[j] = tv[j - 1]; tv[j - 1] = a;
                    int   c = tix[j]; tix[j] = tix[j - 1]; tix[j - 1] = c;
                }
            }
        }
    }

    // merge across 32 lanes: 8 rounds of wave argmax + pop
    float wv[8]; int wi[8];
#pragma unroll
    for (int j = 0; j < 8; ++j) {
        float bv = tv[0];
        int   bl = lane;
#pragma unroll
        for (int off = 16; off > 0; off >>= 1) {
            const float ov = __shfl_xor(bv, off, 32);
            const int   ol = __shfl_xor(bl, off, 32);
            if (ov > bv || (ov == bv && ol < bl)) { bv = ov; bl = ol; }
        }
        wv[j] = bv;
        wi[j] = __shfl(tix[0], bl, 32);
        if (lane == bl) {
#pragma unroll
            for (int q = 0; q < 7; ++q) { tv[q] = tv[q + 1]; tix[q] = tix[q + 1]; }
            tv[7] = NEG_INF;
        }
    }

    // Z_full for the reference's 1e-6 renormalization term
    const float mx = wv[0];
    float zs = 0.f;
    for (int i = lane; i < NNODE; i += 32) zs += __expf(srow[i] - mx);
#pragma unroll
    for (int off = 16; off > 0; off >>= 1) zs += __shfl_xor(zs, off, 32);

    float p[8]; float s8 = 0.f;
#pragma unroll
    for (int j = 0; j < 8; ++j) { p[j] = __expf(wv[j] - mx); s8 += p[j]; }
    const float denom = s8 + 1e-6f * zs;

    if (lane == 0) {
#pragma unroll
        for (int j = 0; j < 8; ++j) {
            top_idx[(size_t)g * 8 + j] = wi[j];
            top_val[(size_t)g * 8 + j] = p[j] / denom;
        }
    }
}

// ---------------------------------------------------------------------------
// Kernel 2: scores strip (16 x 4096) via bf16 WMMA into 256 KB dynamic LDS,
// then top-8 + softmax per row -> (idx,val) lists in ws.
// ---------------------------------------------------------------------------
__global__ void scores_topk_kernel(const __bf16* __restrict__ qb,
                                   const __bf16* __restrict__ kb,
                                   int* __restrict__ top_idx,
                                   float* __restrict__ top_val)
{
    extern __shared__ float sc[];            // [16][4096] f32 = 256 KB
    const int row0 = blockIdx.x * 16;        // flattened B*N row
    const int b    = row0 >> 12;             // batch (N = 4096)
    const int lane = threadIdx.x & 31;
    const int wave = threadIdx.x >> 5;
    const int m    = lane & 15;
    const int khalf = (lane >> 4) * 8;
    const int dh    = (lane >> 4) * 8;
    const int mof   = (lane >> 4) * 8;

    // A fragments: 16 q rows, K = 256 -> 8 fragments held in registers
    v16bf afr[8];
    const __bf16* qrow = qb + (size_t)(row0 + m) * HIDDEN;
#pragma unroll
    for (int kt = 0; kt < 8; ++kt) {
        const int k0 = kt * 32 + khalf;
#pragma unroll
        for (int h = 0; h < 8; ++h) afr[kt][h]     = qrow[k0 + h];
#pragma unroll
        for (int h = 0; h < 8; ++h) afr[kt][8 + h] = qrow[k0 + 16 + h];
    }

    const __bf16* kbatch = kb + (size_t)b * NNODE * HIDDEN;
    for (int ct = wave; ct < NNODE / 16; ct += 8) {
        const __bf16* krow = kbatch + (size_t)(ct * 16 + m) * HIDDEN;
        __builtin_prefetch(krow + 16 * HIDDEN, 0, 1);   // global_prefetch_b8
        v8f acc = {};
#pragma unroll
        for (int kt = 0; kt < 8; ++kt) {
            v16bf bfr;
#pragma unroll
            for (int h = 0; h < 16; ++h)
                bfr[h] = krow[kt * 32 + dh + (h & 7) + ((h >> 3) << 4)];
            acc = __builtin_amdgcn_wmma_f32_16x16x32_bf16(
                false, afr[kt], false, bfr, (short)0, acc, false, false);
        }
        const int col = ct * 16 + m;
#pragma unroll
        for (int v = 0; v < 8; ++v)
            sc[(size_t)(v + mof) * NNODE + col] = acc[v] * 0.0625f; // 1/sqrt(256)
    }
    __syncthreads();

    for (int rr = 0; rr < 2; ++rr) {
        const int r = wave + rr * 8;               // 8 waves x 2 = 16 rows
        topk_row(sc + (size_t)r * NNODE, lane, row0 + r, top_idx, top_val);
    }
}

// ---------------------------------------------------------------------------
// Kernel 3: non-temporal zero fill of the 268 MB output.
// ---------------------------------------------------------------------------
__global__ void zero_kernel(f32x4* __restrict__ out4)
{
    const size_t tid = (size_t)blockIdx.x * blockDim.x + threadIdx.x;
    const size_t stride = (size_t)gridDim.x * blockDim.x;   // 4,194,304
    const f32x4 z = {0.f, 0.f, 0.f, 0.f};
#pragma unroll
    for (int i = 0; i < 4; ++i)
        __builtin_nontemporal_store(z, &out4[tid + (size_t)i * stride]);
}

// ---------------------------------------------------------------------------
// Kernel 4: fused mask + symmetrize scatter: out[r][c]+=v/2, out[c][r]+=v/2.
// ---------------------------------------------------------------------------
__global__ void scatter_kernel(const int* __restrict__ top_idx,
                               const float* __restrict__ top_val,
                               float* __restrict__ out)
{
    const int t = blockIdx.x * 256 + threadIdx.x;     // 131072 total
    const int g = t >> 3;                             // flattened row
    const int b = g >> 12;
    const int r = g & (NNODE - 1);
    const int c = top_idx[t];
    const float v = 0.5f * top_val[t];
    float* base = out + (size_t)b * NNODE * NNODE;
    __hip_atomic_fetch_add(&base[(size_t)r * NNODE + c], v,
                           __ATOMIC_RELAXED, __HIP_MEMORY_SCOPE_AGENT);
    __hip_atomic_fetch_add(&base[(size_t)c * NNODE + r], v,
                           __ATOMIC_RELAXED, __HIP_MEMORY_SCOPE_AGENT);
}

// ---------------------------------------------------------------------------
extern "C" void kernel_launch(void* const* d_in, const int* in_sizes, int n_in,
                              void* d_out, int out_size, void* d_ws, size_t ws_size,
                              hipStream_t stream)
{
    const float* X  = (const float*)d_in[0];   // [4,4096,256]
    const float* Wq = (const float*)d_in[1];   // [256,256]
    const float* bq = (const float*)d_in[2];   // [256]
    const float* Wk = (const float*)d_in[3];   // [256,256]
    const float* bk = (const float*)d_in[4];   // [256]
    float* out = (float*)d_out;                // [4,4096,4096]

    char* ws = (char*)d_ws;
    __bf16* qb = (__bf16*)ws;                                  // 8 MiB
    __bf16* kb = (__bf16*)(ws + ((size_t)8 << 20));            // 8 MiB
    int*    ti = (int*)  (ws + ((size_t)16 << 20));            // 512 KiB
    float*  tv = (float*)(ws + ((size_t)16 << 20) + ((size_t)512 << 10));

    const int strips = NBATCH * NNODE / 16;    // 1024

    proj_kernel<<<strips, 256, 0, stream>>>(X, Wq, bq, Wk, bk, qb, kb);
    scores_topk_kernel<<<strips, 256, 16 * NNODE * sizeof(float), stream>>>(qb, kb, ti, tv);
    zero_kernel<<<16384, 256, 0, stream>>>((f32x4*)out);
    scatter_kernel<<<(NBATCH * NNODE * 8) / 256, 256, 0, stream>>>(ti, tv, out);
}
